// WindowAttention_16234976379244
// MI455X (gfx1250) — matrix operand order repaired and checked by reading, not compiled, not run on hardware
//
#include <hip/hip_runtime.h>

typedef __bf16 bf16;
typedef bf16  v16bf __attribute__((ext_vector_type(16)));
typedef bf16  v8bf  __attribute__((ext_vector_type(8)));
typedef bf16  v4bf  __attribute__((ext_vector_type(4)));
typedef float v8f   __attribute__((ext_vector_type(8)));

#define NWIN 49
#define NP   64
#define DIMC 128
#define NH   4
#define HD   32
#define ATTN_SCALE 0.17677669529663687f  // 32^-0.5

__device__ __forceinline__ v8f wmma_bf16(v16bf a, v16bf b, v8f c) {
  // D = A(16x32 bf16) * B(32x16 bf16) + C(16x16 f32)
  return __builtin_amdgcn_wmma_f32_16x16x32_bf16(
      /*neg_a=*/false, a, /*neg_b=*/false, b,
      /*c_mod=*/(short)0, c, /*reuse_a=*/false, /*reuse_b=*/false);
}

// A fragment 16x32 (MxK) bf16. `row` = &A[m0 + (lane&15)][0], K contiguous.
// ISA: lanes 0-15 hold K = k0+{0..7,16..23}; lanes 16-31 hold K = k0+{8..15,24..31}.
__device__ __forceinline__ v16bf load_a_frag(const bf16* row, int k0, int lane) {
  const int koff = (lane & 16) ? 8 : 0;
  union { v16bf v; v8bf h[2]; } u;
  u.h[0] = *(const v8bf*)(row + k0 + koff);
  u.h[1] = *(const v8bf*)(row + k0 + 16 + koff);
  return u.v;
}

// B fragment 32x16 (KxN) bf16. `col` = &B_T[n0 + (lane&15)][0], K contiguous per fixed N.
// ISA: lanes 0-15 hold K = k0+{0..7,16..23}; lanes 16-31 hold K = k0+{8..15,24..31}.
__device__ __forceinline__ v16bf load_b_frag(const bf16* col, int k0, int lane) {
  const int koff = (lane & 16) ? 8 : 0;
  union { v16bf v; v8bf h[2]; } u;
  u.h[0] = *(const v8bf*)(col + k0 + koff);
  u.h[1] = *(const v8bf*)(col + k0 + 16 + koff);
  return u.v;
}

// ---------------------------------------------------------------------------
// Pre-pass: convert qkv_w [384x128] and proj_w [128x128] f32 -> bf16 scratch.
// ---------------------------------------------------------------------------
__global__ void convert_weights_kernel(const float* __restrict__ qkv_w,
                                       const float* __restrict__ proj_w,
                                       bf16* __restrict__ wq,
                                       bf16* __restrict__ wp) {
  int i = blockIdx.x * blockDim.x + threadIdx.x;
  if (i < 3 * DIMC * DIMC) wq[i] = (bf16)qkv_w[i];
  if (i < DIMC * DIMC)     wp[i] = (bf16)proj_w[i];
}

// ---------------------------------------------------------------------------
// Fused window attention: one workgroup (4 wave32) per window.
// LDS plan (dynamic, 73728 B; offsets in bf16 elements):
//   [0    ) xs  64x128           (aliased as AO after QKV)
//   [8192 ) Qs  [4][64][32]
//   [16384) Ks  [4][64][32]
//   [24576) Vts [4][32][64]      (V transposed: key contiguous)
//   [32768) Ps  [4 waves][16][64]
// ---------------------------------------------------------------------------
__global__ __launch_bounds__(128, 1) void win_attn_kernel(
    const float* __restrict__ x,
    const float* __restrict__ qkv_b,
    const float* __restrict__ rel_table,
    const float* __restrict__ proj_b,
    const int*   __restrict__ rel_index,
    const bf16*  __restrict__ wqkv,
    const bf16*  __restrict__ wproj,
    float* __restrict__ out) {
  extern __shared__ __align__(16) char smem_raw[];
  bf16* sm  = (bf16*)smem_raw;
  bf16* xs  = sm;            // 64x128
  bf16* Qs  = sm + 8192;     // [4][64][32]
  bf16* Ks  = sm + 16384;    // [4][64][32]
  bf16* Vts = sm + 24576;    // [4][32][64]
  bf16* Ps  = sm + 32768;    // [4][16][64]
  bf16* AOs = xs;            // alias: xs dead after post-QKV barrier

  const int tid  = threadIdx.x;
  // wave id is uniform within a wave: force it (and all derived control/address
  // math) into SGPRs so q/k/v store selection is branchless scalar code.
  const int wave = __builtin_amdgcn_readfirstlane(tid >> 5);
  const int lane = tid & 31;
  const int lm   = lane & 15;         // N (or M-row) index within a tile
  const int mrow = (lane >> 4) << 3;  // C-fragment: lanes>=16 hold rows m+8..m+15
  const int bn   = blockIdx.x;

  const float* xw = x + (size_t)bn * NWIN * DIMC;

  // Weights are L2-resident; pull them toward the WGP.
  __builtin_prefetch(wqkv + (tid << 6), 0, 1);
  __builtin_prefetch(wproj + (tid << 5), 0, 1);

  // ---- stage x -> LDS bf16, vectorized (b128 load -> packed b64 store) ---
  {
    const float4* xw4 = (const float4*)xw;          // 1568 float4 per window
    for (int i = tid; i < (NWIN * DIMC) / 4; i += 128) {
      const float4 v = xw4[i];
      v4bf p;
      p.x = (bf16)v.x; p.y = (bf16)v.y; p.z = (bf16)v.z; p.w = (bf16)v.w;
      *(v4bf*)(xs + 4 * i) = p;
    }
    const v4bf z4 = {};
    for (int i = (NWIN * DIMC) / 4 + tid; i < (NP * DIMC) / 4; i += 128)
      *(v4bf*)(xs + 4 * i) = z4;                    // zero-pad rows 49..63
  }
  __syncthreads();

  // ---- QKV GEMM: qkv[64x384] = xs[64x128] * wqkv^T -----------------------
  // Wave owns N-tiles [wave*6, wave*6+6); 4 k-steps per tile.
  // unroll 3 -> three independent WMMA chains to fill RAW-hazard slots.
  for (int mt = 0; mt < 4; ++mt) {
    const bf16* arow = xs + (mt * 16 + lm) * DIMC;
    const v16bf a0 = load_a_frag(arow, 0, lane);
    const v16bf a1 = load_a_frag(arow, 32, lane);
    const v16bf a2 = load_a_frag(arow, 64, lane);
    const v16bf a3 = load_a_frag(arow, 96, lane);
#pragma unroll 3
    for (int j = 0; j < 6; ++j) {
      const int nt   = wave * 6 + j;       // scalar
      const int cout = nt * 16 + lm;
      const bf16* bcol = wqkv + (size_t)cout * DIMC;
      v8f acc = {};
      acc = wmma_bf16(a0, load_b_frag(bcol, 0, lane), acc);
      acc = wmma_bf16(a1, load_b_frag(bcol, 32, lane), acc);
      acc = wmma_bf16(a2, load_b_frag(bcol, 64, lane), acc);
      acc = wmma_bf16(a3, load_b_frag(bcol, 96, lane), acc);
      const float bias = qkv_b[cout];
      // Branchless scatter: Qs/Ks/Vts are contiguous; select base + strides.
      const int sect = nt >> 3;            // 0=q 1=k 2=v  (SGPR)
      const int head = (nt & 7) >> 1;      // SGPR
      const int isV  = (sect == 2);
      const int sn   = isV ? 1 : HD;       // token-index stride
      const int sh   = isV ? NP : 1;       // head-dim stride
      const int hd   = ((nt & 1) << 4) + lm;
      bf16* dst = sm + 8192 * (1 + sect) + head * (NP * HD)
                + (mt * 16 + mrow) * sn + hd * sh;
#pragma unroll
      for (int g = 0; g < 8; ++g) dst[g * sn] = (bf16)(acc[g] + bias);
    }
  }
  __syncthreads();

  // ---- attention: wave == head -------------------------------------------
  {
    const int h = wave;
    const bf16* Qh = Qs + h * (NP * HD);
    const bf16* Kh = Ks + h * (NP * HD);
    const bf16* Vh = Vts + h * (HD * NP);
    bf16* Pw = Ps + wave * (16 * NP);

    for (int mt = 0; mt < 4; ++mt) {
      const v16bf qa = load_a_frag(Qh + (mt * 16 + lm) * HD, 0, lane);
      v8f s[4];
#pragma unroll
      for (int nt = 0; nt < 4; ++nt) {
        v8f z = {};
        // K^T as B: N = key, K = head-dim (contiguous in Ks rows).
        s[nt] = wmma_bf16(qa, load_b_frag(Kh + (nt * 16 + lm) * HD, 0, lane), z);
      }
      // scale + relative-position bias (clamped gather) + key-padding mask
#pragma unroll
      for (int nt = 0; nt < 4; ++nt) {
        const int kc  = nt * 16 + lm;
        const int kcc = kc < (NWIN - 1) ? kc : (NWIN - 1);
#pragma unroll
        for (int g = 0; g < 8; ++g) {
          const int q  = mt * 16 + mrow + g;
          const int qc = q < (NWIN - 1) ? q : (NWIN - 1);
          const float b = rel_table[rel_index[qc * NWIN + kcc] * NH + h];
          const float val = fmaf(s[nt][g], ATTN_SCALE, b);
          s[nt][g] = (kc < NWIN) ? val : -1.0e30f;
        }
      }
      // row softmax: a row lives in one 16-lane half (xor 1,2,4,8), 4 tiles wide
#pragma unroll
      for (int g = 0; g < 8; ++g) {
        float m = fmaxf(fmaxf(s[0][g], s[1][g]), fmaxf(s[2][g], s[3][g]));
#pragma unroll
        for (int d = 1; d < 16; d <<= 1) m = fmaxf(m, __shfl_xor(m, d, 32));
        const float e0 = __expf(s[0][g] - m);
        const float e1 = __expf(s[1][g] - m);
        const float e2 = __expf(s[2][g] - m);
        const float e3 = __expf(s[3][g] - m);
        float sum = e0 + e1 + e2 + e3;
#pragma unroll
        for (int d = 1; d < 16; d <<= 1) sum += __shfl_xor(sum, d, 32);
        const float r = __builtin_amdgcn_rcpf(sum);
        s[0][g] = e0 * r; s[1][g] = e1 * r; s[2][g] = e2 * r; s[3][g] = e3 * r;
      }
      // P -> bf16 in wave-private LDS (C-layout -> row-major)
#pragma unroll
      for (int nt = 0; nt < 4; ++nt)
#pragma unroll
        for (int g = 0; g < 8; ++g)
          Pw[(mrow + g) * NP + nt * 16 + lm] = (bf16)s[nt][g];
      // PV: out[16x32] = P[16x64] * V[64x32]; V as B via Vts (key contiguous)
      const v16bf pa0 = load_a_frag(Pw + lm * NP, 0, lane);
      const v16bf pa1 = load_a_frag(Pw + lm * NP, 32, lane);
#pragma unroll
      for (int nt = 0; nt < 2; ++nt) {
        const bf16* vcol = Vh + (nt * 16 + lm) * NP;
        v8f o = {};
        o = wmma_bf16(pa0, load_b_frag(vcol, 0, lane), o);
        o = wmma_bf16(pa1, load_b_frag(vcol, 32, lane), o);
#pragma unroll
        for (int g = 0; g < 8; ++g) {
          const int n = mt * 16 + mrow + g;
          AOs[n * DIMC + h * HD + nt * 16 + lm] = (bf16)o[g];
        }
      }
    }
  }
  __syncthreads();

  // ---- proj GEMM: out[64x128] = AO[64x128] * wproj^T + proj_b ------------
  // mt fully unrolled: the n<NWIN tail predicate constant-folds for mt 0..2.
#pragma unroll
  for (int mt = 0; mt < 4; ++mt) {
    const bf16* arow = AOs + (mt * 16 + lm) * DIMC;
    const v16bf a0 = load_a_frag(arow, 0, lane);
    const v16bf a1 = load_a_frag(arow, 32, lane);
    const v16bf a2 = load_a_frag(arow, 64, lane);
    const v16bf a3 = load_a_frag(arow, 96, lane);
#pragma unroll
    for (int j = 0; j < 2; ++j) {
      const int nt   = wave * 2 + j;       // scalar
      const int cout = nt * 16 + lm;
      const bf16* bcol = wproj + (size_t)cout * DIMC;
      v8f acc = {};
      acc = wmma_bf16(a0, load_b_frag(bcol, 0, lane), acc);
      acc = wmma_bf16(a1, load_b_frag(bcol, 32, lane), acc);
      acc = wmma_bf16(a2, load_b_frag(bcol, 64, lane), acc);
      acc = wmma_bf16(a3, load_b_frag(bcol, 96, lane), acc);
      const float bias = proj_b[cout];
#pragma unroll
      for (int g = 0; g < 8; ++g) {
        const int n = mt * 16 + mrow + g;
        if (n < NWIN)
          out[((size_t)bn * NWIN + n) * DIMC + cout] = acc[g] + bias;
      }
    }
  }
}

extern "C" void kernel_launch(void* const* d_in, const int* in_sizes, int n_in,
                              void* d_out, int out_size, void* d_ws, size_t ws_size,
                              hipStream_t stream) {
  const float* x         = (const float*)d_in[0];
  const float* qkv_w     = (const float*)d_in[1];
  const float* qkv_b     = (const float*)d_in[2];
  const float* rel_table = (const float*)d_in[3];
  const float* proj_w    = (const float*)d_in[4];
  const float* proj_b    = (const float*)d_in[5];
  const int*   rel_index = (const int*)d_in[6];

  bf16* wq = (bf16*)d_ws;                 // 384*128 bf16
  bf16* wp = wq + 3 * DIMC * DIMC;        // 128*128 bf16 (total 128 KB <= ws_size)

  const int BN = in_sizes[0] / (NWIN * DIMC);

  convert_weights_kernel<<<(3 * DIMC * DIMC + 255) / 256, 256, 0, stream>>>(
      qkv_w, proj_w, wq, wp);

  const size_t lds_bytes = 73728;  // 72 KB dynamic LDS (320 KB/WGP available)
  (void)hipFuncSetAttribute((const void*)win_attn_kernel,
                            hipFuncAttributeMaxDynamicSharedMemorySize,
                            (int)lds_bytes);
  win_attn_kernel<<<BN, 128, lds_bytes, stream>>>(
      x, qkv_b, rel_table, proj_b, rel_index, wq, wp, (float*)d_out);
}
